// RobustMambaDiagSSM_40896678592897
// MI455X (gfx1250) — matrix operand order repaired and checked by reading, not compiled, not run on hardware
//
#include <hip/hip_runtime.h>
#include <hip/hip_bf16.h>
#include <math.h>

typedef __bf16 bf16;
typedef __attribute__((ext_vector_type(16))) __bf16 v16bf;
typedef __attribute__((ext_vector_type(8)))  __bf16 v8bf;
typedef __attribute__((ext_vector_type(8)))  float  v8f;

#define B_SZ   4
#define T_SZ   2048
#define D_SZ   1024
#define N_SZ   1024
#define DOUT   1024
#define BT     (B_SZ * T_SZ)          // 8192 rows
#define NCHUNK 16
#define CLEN   (T_SZ / NCHUNK)        // 128
#define EPS_A  1e-4f

__device__ __forceinline__ float softplus_f(float x) {
    return x > 20.0f ? x : log1pf(expf(x));
}

// ---------------------------------------------------------------------------
// Kernel 1: spectral norm via power iteration. grid=2 (block 0: W_in, 1: W_out)
// Both matrices are 1024x1024. One 1024-thread workgroup each.
// ---------------------------------------------------------------------------
__global__ __launch_bounds__(1024) void sigma_kernel(const float* __restrict__ Win,
                                                     const float* __restrict__ Wout,
                                                     float* __restrict__ sig) {
    const float* W = (blockIdx.x == 0) ? Win : Wout;
    __shared__ float v[1024];
    __shared__ float y[1024];
    __shared__ float red[1024];
    const int tid = threadIdx.x;
    v[tid] = 1.0f;
    __syncthreads();
    for (int it = 0; it < 14; ++it) {
        // normalize v
        red[tid] = v[tid] * v[tid];
        __syncthreads();
        for (int o = 512; o > 0; o >>= 1) { if (tid < o) red[tid] += red[tid + o]; __syncthreads(); }
        const float inv = rsqrtf(fmaxf(red[0], 1e-30f));
        __syncthreads();
        v[tid] *= inv;
        __syncthreads();
        // y = W v  (thread = row)
        {
            const float* Wr = W + (size_t)tid * 1024;
            float acc = 0.0f;
            for (int k = 0; k < 1024; k += 4) {
                acc = fmaf(Wr[k],   v[k],   acc);
                acc = fmaf(Wr[k+1], v[k+1], acc);
                acc = fmaf(Wr[k+2], v[k+2], acc);
                acc = fmaf(Wr[k+3], v[k+3], acc);
            }
            y[tid] = acc;
        }
        __syncthreads();
        // v = W^T y  (thread = column; lane-consecutive columns -> coalesced)
        {
            float acc = 0.0f;
            for (int k = 0; k < 1024; ++k) acc = fmaf(W[(size_t)k * 1024 + tid], y[k], acc);
            v[tid] = acc;
        }
        __syncthreads();
    }
    // final normalize, y = W v, sigma = ||y||
    red[tid] = v[tid] * v[tid];
    __syncthreads();
    for (int o = 512; o > 0; o >>= 1) { if (tid < o) red[tid] += red[tid + o]; __syncthreads(); }
    const float inv = rsqrtf(fmaxf(red[0], 1e-30f));
    __syncthreads();
    v[tid] *= inv;
    __syncthreads();
    {
        const float* Wr = W + (size_t)tid * 1024;
        float acc = 0.0f;
        for (int k = 0; k < 1024; ++k) acc = fmaf(Wr[k], v[k], acc);
        red[tid] = acc * acc;
    }
    __syncthreads();
    for (int o = 512; o > 0; o >>= 1) { if (tid < o) red[tid] += red[tid + o]; __syncthreads(); }
    if (tid == 0) sig[blockIdx.x] = sqrtf(fmaxf(red[0], 0.0f));
}

// ---------------------------------------------------------------------------
// Fragment helpers (layouts per CDNA5 ISA §7.12.2, 16-bit A 16x32 /
// 16-bit B 32x16). lm = lane&15, kh = lane>>4.
// ---------------------------------------------------------------------------
__device__ __forceinline__ v16bf load_a_frag(const bf16* As, int row, int kh) {
    const v8bf lo = *(const v8bf*)&As[row * 32 + kh * 8];
    const v8bf hi = *(const v8bf*)&As[row * 32 + 16 + kh * 8];
    return __builtin_shufflevector(lo, hi, 0,1,2,3,4,5,6,7,8,9,10,11,12,13,14,15);
}
__device__ __forceinline__ v16bf load_b_frag(const bf16* Bs, int nrow, int kh) {
    const v8bf lo = *(const v8bf*)&Bs[nrow * 32 + kh * 8];
    const v8bf hi = *(const v8bf*)&Bs[nrow * 32 + 16 + kh * 8];
    return __builtin_shufflevector(lo, hi, 0,1,2,3,4,5,6,7,8,9,10,11,12,13,14,15);
}

// ---------------------------------------------------------------------------
// Kernel 2: fused GEMM (u @ [W_in; pn_delta; pn_b; pn_c]^T for one n-range)
// + full activation/2x2-spectral epilogue. Writes a, b*u_scaled, c.
// Block = 256 thr (8 waves). Tile: M=128, 4 x 16 n-cols, K-step 32 (bf16 WMMA).
// ---------------------------------------------------------------------------
__global__ __launch_bounds__(256) void gemm_act_kernel(
    const float* __restrict__ u,   const float* __restrict__ Win,
    const float* __restrict__ pnw, const float* __restrict__ pnb,
    const float* __restrict__ alpha_log, const float* __restrict__ delta_bias,
    const float* __restrict__ log_gamma, const float* __restrict__ sig,
    float* __restrict__ Abuf, float* __restrict__ BUbuf, float* __restrict__ Cbuf)
{
    __shared__ __align__(32) bf16 As[128 * 32];
    __shared__ __align__(32) bf16 Bs[64 * 32];
    const int m0 = blockIdx.x * 128;
    const int n0 = blockIdx.y * 16;
    const int tid  = threadIdx.x;
    const int lane = tid & 31, wave = tid >> 5;
    const int lm = lane & 15, kh = lane >> 4;

    v8f accA = {}, accD = {}, accB = {}, accC = {};

    for (int k0 = 0; k0 < D_SZ; k0 += 32) {
        __syncthreads();
        // stage A tile 128x32 (fp32 -> bf16)
        {
            const int r = tid >> 3, c = (tid & 7) * 4;
            if (k0 + 32 < D_SZ)
                __builtin_prefetch(u + (size_t)(m0 + r) * D_SZ + (k0 + 32) + c, 0, 0);
            #pragma unroll
            for (int rr = 0; rr < 128; rr += 32) {
                const float4 f = *(const float4*)(u + (size_t)(m0 + r + rr) * D_SZ + k0 + c);
                bf16* dst = &As[(r + rr) * 32 + c];
                dst[0] = (bf16)f.x; dst[1] = (bf16)f.y; dst[2] = (bf16)f.z; dst[3] = (bf16)f.w;
            }
        }
        // stage B tile 64x32: rows 0-15 W_in[n0+i], 16-31 pn_delta, 32-47 pn_b, 48-63 pn_c
        {
            const int r = tid >> 2, c = (tid & 3) * 8;
            const int i = r & 15, blk = r >> 4;
            const float* src = (blk == 0)
                ? Win + (size_t)(n0 + i) * D_SZ + k0
                : pnw + (size_t)((blk - 1) * N_SZ + n0 + i) * D_SZ + k0;
            const float4 f0 = *(const float4*)(src + c);
            const float4 f1 = *(const float4*)(src + c + 4);
            bf16* dst = &Bs[r * 32 + c];
            dst[0] = (bf16)f0.x; dst[1] = (bf16)f0.y; dst[2] = (bf16)f0.z; dst[3] = (bf16)f0.w;
            dst[4] = (bf16)f1.x; dst[5] = (bf16)f1.y; dst[6] = (bf16)f1.z; dst[7] = (bf16)f1.w;
        }
        __syncthreads();
        const v16bf af = load_a_frag(As, wave * 16 + lm, kh);
        const v16bf b0 = load_b_frag(Bs, 0 * 16 + lm, kh);
        const v16bf b1 = load_b_frag(Bs, 1 * 16 + lm, kh);
        const v16bf b2 = load_b_frag(Bs, 2 * 16 + lm, kh);
        const v16bf b3 = load_b_frag(Bs, 3 * 16 + lm, kh);
        accA = __builtin_amdgcn_wmma_f32_16x16x32_bf16(false, af, false, b0, (short)0, accA, false, false);
        accD = __builtin_amdgcn_wmma_f32_16x16x32_bf16(false, af, false, b1, (short)0, accD, false, false);
        accB = __builtin_amdgcn_wmma_f32_16x16x32_bf16(false, af, false, b2, (short)0, accB, false, false);
        accC = __builtin_amdgcn_wmma_f32_16x16x32_bf16(false, af, false, b3, (short)0, accC, false, false);
    }

    // ---- fused activation epilogue ----
    const float g    = expf(log_gamma[0]);
    const float gs   = g / fmaxf(sig[0], 1.0f);          // bound=1, so scale = max(sigma,1)
    const int   n    = n0 + lm;
    const float alph = softplus_f(alpha_log[n]);
    const float db   = delta_bias[n];
    const float pbd  = pnb[n];
    const float pbb  = pnb[N_SZ + n];
    const float pbc  = pnb[2 * N_SZ + n];
    #pragma unroll
    for (int r = 0; r < 8; ++r) {
        const int   m     = m0 + wave * 16 + kh * 8 + r;   // C/D layout: VGPR r -> M = r (+8 for hi lanes)
        const float us    = accA[r] * gs;
        const float delta = softplus_f(accD[r] + pbd + db);
        float a = fminf(expf(-delta * alph), 1.0f - EPS_A);
        float b = tanhf(accB[r] + pbb);
        float c = tanhf(accC[r] + pbc);
        // exact spectral norm of [[a,b],[c,0]]
        const float p    = a * a + c * c;
        const float rr   = b * b;
        const float q    = a * b;
        const float disc = (p - rr) * (p - rr) + 4.0f * q * q;
        const float lam  = 0.5f * (p + rr + sqrtf(disc + 1e-12f));
        const float sgm  = sqrtf(lam + 1e-12f);
        const float is   = 1.0f / fmaxf(sgm, 1.0f);
        a *= is; b *= is; c *= is;
        const size_t o = (size_t)m * N_SZ + n;
        Abuf[o]  = a;
        BUbuf[o] = b * us;
        Cbuf[o]  = c;
    }
}

// ---------------------------------------------------------------------------
// 3-pass chunked associative scan: z_{t+1} = a_t z_t + bu_t ; y_hat_t = c_t z_t
// ---------------------------------------------------------------------------
__global__ __launch_bounds__(256) void scan_pass1(const float* __restrict__ A, const float* __restrict__ BU,
                                                  float* __restrict__ Ag, float* __restrict__ Bg) {
    const int idx = blockIdx.x * 256 + threadIdx.x;      // 65536 threads
    const int n = idx & (N_SZ - 1);
    const int b = (idx >> 10) & (B_SZ - 1);
    const int j = idx >> 12;                             // chunk 0..15
    size_t base = ((size_t)b * T_SZ + (size_t)j * CLEN) * N_SZ + n;
    float ap = 1.0f, z = 0.0f;
    for (int t = 0; t < CLEN; ++t) {
        const float a  = A[base];
        const float bu = BU[base];
        z  = fmaf(a, z, bu);
        ap *= a;
        base += N_SZ;
    }
    const int ch = b * N_SZ + n;
    Ag[(size_t)j * (B_SZ * N_SZ) + ch] = ap;
    Bg[(size_t)j * (B_SZ * N_SZ) + ch] = z;
}

__global__ __launch_bounds__(256) void scan_pass2(const float* __restrict__ z0, const float* __restrict__ Ag,
                                                  const float* __restrict__ Bg, float* __restrict__ Zs) {
    const int ch = blockIdx.x * 256 + threadIdx.x;       // 4096 channels
    float z = z0[ch];
    #pragma unroll
    for (int j = 0; j < NCHUNK; ++j) {
        const size_t o = (size_t)j * (B_SZ * N_SZ) + ch;
        Zs[o] = z;                                        // state ENTERING chunk j
        z = fmaf(Ag[o], z, Bg[o]);
    }
}

__global__ __launch_bounds__(256) void scan_pass3(const float* __restrict__ A, const float* __restrict__ BU,
                                                  const float* __restrict__ Zs, float* __restrict__ C) {
    const int idx = blockIdx.x * 256 + threadIdx.x;
    const int n = idx & (N_SZ - 1);
    const int b = (idx >> 10) & (B_SZ - 1);
    const int j = idx >> 12;
    const int ch = b * N_SZ + n;
    float z = Zs[(size_t)j * (B_SZ * N_SZ) + ch];
    size_t base = ((size_t)b * T_SZ + (size_t)j * CLEN) * N_SZ + n;
    for (int t = 0; t < CLEN; ++t) {
        const float a  = A[base];
        const float bu = BU[base];
        const float c  = C[base];
        C[base] = c * z;                 // y_hat_t = c_t * z_t (pre-update state)
        z = fmaf(a, z, bu);
        base += N_SZ;
    }
}

// ---------------------------------------------------------------------------
// Kernel 4: y = y_hat @ (W_out/scale_out)^T. Tile M=128 x N=64, 8 waves,
// each wave: one 16-row strip x 4 n-subtiles.
// ---------------------------------------------------------------------------
__global__ __launch_bounds__(256) void gemm_out_kernel(const float* __restrict__ Y,
                                                       const float* __restrict__ Wout,
                                                       const float* __restrict__ sig,
                                                       float* __restrict__ out) {
    __shared__ __align__(32) bf16 As[128 * 32];
    __shared__ __align__(32) bf16 Bs[64 * 32];
    const int m0 = blockIdx.x * 128;
    const int n0 = blockIdx.y * 64;
    const int tid  = threadIdx.x;
    const int lane = tid & 31, wave = tid >> 5;
    const int lm = lane & 15, kh = lane >> 4;

    v8f acc[4] = {};

    for (int k0 = 0; k0 < N_SZ; k0 += 32) {
        __syncthreads();
        {
            const int r = tid >> 3, c = (tid & 7) * 4;
            if (k0 + 32 < N_SZ)
                __builtin_prefetch(Y + (size_t)(m0 + r) * N_SZ + (k0 + 32) + c, 0, 0);
            #pragma unroll
            for (int rr = 0; rr < 128; rr += 32) {
                const float4 f = *(const float4*)(Y + (size_t)(m0 + r + rr) * N_SZ + k0 + c);
                bf16* dst = &As[(r + rr) * 32 + c];
                dst[0] = (bf16)f.x; dst[1] = (bf16)f.y; dst[2] = (bf16)f.z; dst[3] = (bf16)f.w;
            }
        }
        {
            const int r = tid >> 2, c = (tid & 3) * 8;
            const float* src = Wout + (size_t)(n0 + r) * N_SZ + k0;
            const float4 f0 = *(const float4*)(src + c);
            const float4 f1 = *(const float4*)(src + c + 4);
            bf16* dst = &Bs[r * 32 + c];
            dst[0] = (bf16)f0.x; dst[1] = (bf16)f0.y; dst[2] = (bf16)f0.z; dst[3] = (bf16)f0.w;
            dst[4] = (bf16)f1.x; dst[5] = (bf16)f1.y; dst[6] = (bf16)f1.z; dst[7] = (bf16)f1.w;
        }
        __syncthreads();
        const v16bf af = load_a_frag(As, wave * 16 + lm, kh);
        #pragma unroll
        for (int blk = 0; blk < 4; ++blk) {
            const v16bf bf_ = load_b_frag(Bs, blk * 16 + lm, kh);
            acc[blk] = __builtin_amdgcn_wmma_f32_16x16x32_bf16(false, af, false, bf_, (short)0, acc[blk], false, false);
        }
    }
    const float is = 1.0f / fmaxf(sig[1], 1.0f);
    #pragma unroll
    for (int blk = 0; blk < 4; ++blk) {
        const int n = n0 + blk * 16 + lm;
        #pragma unroll
        for (int r = 0; r < 8; ++r) {
            const int m = m0 + wave * 16 + kh * 8 + r;
            out[(size_t)m * DOUT + n] = acc[blk][r] * is;
        }
    }
}

// ---------------------------------------------------------------------------
extern "C" void kernel_launch(void* const* d_in, const int* in_sizes, int n_in,
                              void* d_out, int out_size, void* d_ws, size_t ws_size,
                              hipStream_t stream) {
    const float* u          = (const float*)d_in[0];
    const float* z0         = (const float*)d_in[1];
    const float* Win        = (const float*)d_in[2];
    const float* Wout       = (const float*)d_in[3];
    const float* pnw        = (const float*)d_in[4];
    const float* pnb        = (const float*)d_in[5];
    const float* alpha_log  = (const float*)d_in[6];
    const float* delta_bias = (const float*)d_in[7];
    const float* log_gamma  = (const float*)d_in[8];

    float* ws    = (float*)d_ws;
    float* sig   = ws;                                    // [0]=sigma_in [1]=sigma_out
    float* Abuf  = ws + 16;
    float* BUbuf = Abuf  + (size_t)BT * N_SZ;
    float* Cbuf  = BUbuf + (size_t)BT * N_SZ;
    float* Ag    = Cbuf  + (size_t)BT * N_SZ;
    float* Bg    = Ag + (size_t)B_SZ * N_SZ * NCHUNK;
    float* Zs    = Bg + (size_t)B_SZ * N_SZ * NCHUNK;

    sigma_kernel<<<2, 1024, 0, stream>>>(Win, Wout, sig);
    gemm_act_kernel<<<dim3(BT / 128, N_SZ / 16), 256, 0, stream>>>(
        u, Win, pnw, pnb, alpha_log, delta_bias, log_gamma, sig, Abuf, BUbuf, Cbuf);
    scan_pass1<<<(B_SZ * N_SZ * NCHUNK) / 256, 256, 0, stream>>>(Abuf, BUbuf, Ag, Bg);
    scan_pass2<<<(B_SZ * N_SZ) / 256, 256, 0, stream>>>(z0, Ag, Bg, Zs);
    scan_pass3<<<(B_SZ * N_SZ * NCHUNK) / 256, 256, 0, stream>>>(Abuf, BUbuf, Zs, Cbuf);
    gemm_out_kernel<<<dim3(BT / 128, DOUT / 64), 256, 0, stream>>>(Cbuf, Wout, sig, (float*)d_out);
}
